// CausalSelfAttention_62989990363647
// MI455X (gfx1250) — matrix operand order repaired and checked by reading, not compiled, not run on hardware
//
#include <hip/hip_runtime.h>
#include <hip/hip_bf16.h>

// B=4, T=2048, C=1024, NH=16, HS=64. BH = B*NH = 64. M = B*T = 8192.

typedef __attribute__((ext_vector_type(16))) __bf16 bf16x16;
typedef __attribute__((ext_vector_type(8)))  float  f32x8;
typedef int v4i_vs __attribute__((vector_size(16)));  // matches builtin param type

union FragU { bf16x16 v; __bf16 e[16]; };

#if defined(__HIP_DEVICE_COMPILE__) &&                                   \
    __has_builtin(__builtin_amdgcn_global_load_async_to_lds_b128) &&     \
    __has_builtin(__builtin_amdgcn_s_wait_asynccnt)
#define USE_ASYNC_LDS 1
#else
#define USE_ASYNC_LDS 0
#endif

// A-matrix fragment (16x32 bf16, M x K), per ISA 7.12.2:
// lanes 0-15: M rows, VGPR p holds K pair; lanes 16-31 same rows, K+8 / K+24.
static __device__ __forceinline__ bf16x16 load_frag_a(const __bf16* base, int stride, int lane) {
  const int m = lane & 15, half = lane >> 4;
  const __bf16* row = base + (size_t)m * stride;
  FragU f;
#pragma unroll
  for (int p = 0; p < 8; ++p) {
    const int kb = ((p & 4) << 2) + half * 8 + ((p & 3) << 1);
    f.e[2 * p]     = row[kb];
    f.e[2 * p + 1] = row[kb + 1];
  }
  return f.v;
}

// B-matrix fragment (32x16 bf16, K x N), source stored N-major rows of K
// (element (k,n) = base[n*stride + k]). Same K order as A fragment:
// lanes 0-15: K 0-7,16-23; lanes 16-31: K 8-15,24-31.
static __device__ __forceinline__ bf16x16 load_frag_b(const __bf16* base, int stride, int lane) {
  const int n = lane & 15, half = lane >> 4;
  const __bf16* row = base + (size_t)n * stride;
  FragU f;
#pragma unroll
  for (int p = 0; p < 8; ++p) {
    const int kb = ((p & 4) << 2) + half * 8 + ((p & 3) << 1);
    f.e[2 * p]     = row[kb];
    f.e[2 * p + 1] = row[kb + 1];
  }
  return f.v;
}

static __device__ __forceinline__ f32x8 wmma_bf16(bf16x16 a, bf16x16 b, f32x8 c) {
  return __builtin_amdgcn_wmma_f32_16x16x32_bf16(false, a, false, b, (short)0, c, false, false);
}

#if USE_ASYNC_LDS
// async copy of 16B (8 bf16) global -> LDS, tracked by ASYNCcnt
static __device__ __forceinline__ void async_cp16(const __bf16* g, __bf16* l) {
  __builtin_amdgcn_global_load_async_to_lds_b128(
      (__attribute__((address_space(1))) v4i_vs*)(__bf16*)g,
      (__attribute__((address_space(3))) v4i_vs*)l, 0, 0);
}
#endif

// ---------------- conversion kernels ----------------
__global__ void f32_to_bf16(const float* __restrict__ src, __bf16* __restrict__ dst, int n) {
  int i = blockIdx.x * blockDim.x + threadIdx.x;
  if (i < n) dst[i] = (__bf16)src[i];
}

// dst[n][k] = src[k][n]  (weights -> N-major-of-K for B fragments)
__global__ void f32_to_bf16_t(const float* __restrict__ src, __bf16* __restrict__ dst,
                              int rows, int cols) {
  int i = blockIdx.x * blockDim.x + threadIdx.x;
  if (i < rows * cols) {
    int k = i / cols, n = i - k * cols;
    dst[(size_t)n * rows + k] = (__bf16)src[i];
  }
}

// ---------------- GEMM: D[M,N] = A[M,K] @ Bt[N,K]^T + bias ----------------
// mode 0: store fp32 to outF.
// mode 1: qkv epilogue -> scatter to flash buffers:
//   part 0 (q -> flash-"K") qk[bh][t][d]
//   part 1 (k -> flash-"Q") kq[bh][t][d] scaled by 0.125*log2(e)
//   part 2 (v)              vt[bh][d][t]
#define BKP 40
__global__ __launch_bounds__(256)
void gemm_wmma(const __bf16* __restrict__ A, const __bf16* __restrict__ Bt,
               const float* __restrict__ bias, float* __restrict__ outF,
               __bf16* __restrict__ qk, __bf16* __restrict__ kq, __bf16* __restrict__ vt,
               int M, int N, int K, int mode) {
  __shared__ __bf16 lA[2][128][BKP];
  __shared__ __bf16 lB[2][128][BKP];
  const int tid  = threadIdx.x;
  const int lane = tid & 31;
  const int wave = tid >> 5;
  const int nbx  = N >> 7;
  const int bx   = blockIdx.x % nbx;
  const int by   = blockIdx.x / nbx;
  const int wm   = (wave >> 1) * 32;   // wave M origin in tile
  const int wn   = (wave & 1) * 64;    // wave N origin in tile
  const int lr   = tid >> 1;           // 0..127 : row loaded by this thread
  const int lc   = (tid & 1) * 16;     // 0/16   : bf16 col offset in K-slab

  f32x8 acc[2][4];
#pragma unroll
  for (int i = 0; i < 2; ++i)
#pragma unroll
    for (int j = 0; j < 4; ++j) acc[i][j] = f32x8{};

  const __bf16* aptr = A  + (size_t)(by * 128 + lr) * K + lc;
  const __bf16* bptr = Bt + (size_t)(bx * 128 + lr) * K + lc;

#if USE_ASYNC_LDS
  // ---- async double-buffered pipeline: ASYNCcnt-tracked global->LDS DMA ----
  {
    async_cp16(aptr,     &lA[0][lr][lc]);
    async_cp16(aptr + 8, &lA[0][lr][lc + 8]);
    async_cp16(bptr,     &lB[0][lr][lc]);
    async_cp16(bptr + 8, &lB[0][lr][lc + 8]);
  }
  for (int k0 = 0; k0 < K; k0 += 32) {
    const int cur = (k0 >> 5) & 1;
    __builtin_amdgcn_s_wait_asynccnt(0);
    __syncthreads();                    // tile[cur] visible to all waves;
                                        // tile[1-cur] reads from prev iter done
    if (k0 + 32 < K) {
      const __bf16* ga = aptr + k0 + 32;
      const __bf16* gb = bptr + k0 + 32;
      async_cp16(ga,     &lA[1 - cur][lr][lc]);
      async_cp16(ga + 8, &lA[1 - cur][lr][lc + 8]);
      async_cp16(gb,     &lB[1 - cur][lr][lc]);
      async_cp16(gb + 8, &lB[1 - cur][lr][lc + 8]);
    }
    bf16x16 af0 = load_frag_a(&lA[cur][wm][0],      BKP, lane);
    bf16x16 af1 = load_frag_a(&lA[cur][wm + 16][0], BKP, lane);
    bf16x16 bfr[4];
#pragma unroll
    for (int j = 0; j < 4; ++j) bfr[j] = load_frag_b(&lB[cur][wn + 16 * j][0], BKP, lane);
#pragma unroll
    for (int j = 0; j < 4; ++j) {
      acc[0][j] = wmma_bf16(af0, bfr[j], acc[0][j]);
      acc[1][j] = wmma_bf16(af1, bfr[j], acc[1][j]);
    }
  }
#else
  // ---- fallback: synchronous staging through VGPRs ----
  for (int k0 = 0; k0 < K; k0 += 32) {
    uint4 av0 = *(const uint4*)(aptr + k0);
    uint4 av1 = *(const uint4*)(aptr + k0 + 8);
    uint4 bv0 = *(const uint4*)(bptr + k0);
    uint4 bv1 = *(const uint4*)(bptr + k0 + 8);
    if (k0 + 32 < K) {
      __builtin_prefetch(aptr + k0 + 32, 0, 1);
      __builtin_prefetch(bptr + k0 + 32, 0, 1);
    }
    __syncthreads();
    *(uint4*)&lA[0][lr][lc]     = av0;
    *(uint4*)&lA[0][lr][lc + 8] = av1;
    *(uint4*)&lB[0][lr][lc]     = bv0;
    *(uint4*)&lB[0][lr][lc + 8] = bv1;
    __syncthreads();

    bf16x16 af0 = load_frag_a(&lA[0][wm][0],      BKP, lane);
    bf16x16 af1 = load_frag_a(&lA[0][wm + 16][0], BKP, lane);
    bf16x16 bfr[4];
#pragma unroll
    for (int j = 0; j < 4; ++j) bfr[j] = load_frag_b(&lB[0][wn + 16 * j][0], BKP, lane);
#pragma unroll
    for (int j = 0; j < 4; ++j) {
      acc[0][j] = wmma_bf16(af0, bfr[j], acc[0][j]);
      acc[1][j] = wmma_bf16(af1, bfr[j], acc[1][j]);
    }
  }
#endif

  const int half = lane >> 4, nn = lane & 15;
#pragma unroll
  for (int i = 0; i < 2; ++i)
#pragma unroll
    for (int j = 0; j < 4; ++j)
#pragma unroll
      for (int g = 0; g < 8; ++g) {
        const int mg = by * 128 + wm + i * 16 + g + half * 8;
        const int ng = bx * 128 + wn + j * 16 + nn;
        float v = acc[i][j][g] + bias[ng];
        if (mode == 0) {
          outF[(size_t)mg * N + ng] = v;
        } else {
          const int b = mg >> 11, t = mg & 2047;
          const int part = ng >> 10, c = ng & 1023;
          const int h = c & 15, d = c >> 4;
          const size_t bh = (size_t)(b * 16 + h);
          if (part == 0)      qk[(bh * 2048 + t) * 64 + d] = (__bf16)v;
          else if (part == 1) kq[(bh * 2048 + t) * 64 + d] = (__bf16)(v * 0.18033688011112042f);
          else                vt[(bh * 64 + d) * 2048 + t] = (__bf16)v;
        }
      }
}

// ---------------- flash attention (Q<->K swapped causal) ----------------
// out y[b, j, d*16+h] = sum_{i<=j} softmax_i( q_i . k_j / 8 ) * v[i,d]
__global__ __launch_bounds__(256)
void flash_attn(const __bf16* __restrict__ qk, const __bf16* __restrict__ kq,
                const __bf16* __restrict__ vt, __bf16* __restrict__ ybuf) {
  __shared__ __bf16 sP[8][16][32];
  const int tid = threadIdx.x, lane = tid & 31, wave = tid >> 5;
  const int bh = blockIdx.x >> 4, jt = blockIdx.x & 15;
  const int b = bh >> 4, h = bh & 15;
  const int j0 = jt * 128 + wave * 16;
  const int half = lane >> 4, nn = lane & 15;

  // flash-Q = pre-scaled k-heads, 16 rows x 64 d -> two A frags
  const __bf16* qbase = kq + ((size_t)bh * 2048 + j0) * 64;
  const bf16x16 aq0 = load_frag_a(qbase,      64, lane);
  const bf16x16 aq1 = load_frag_a(qbase + 32, 64, lane);

  f32x8 accO[4];
#pragma unroll
  for (int t = 0; t < 4; ++t) accO[t] = f32x8{};
  float mst[8], lst[8];
#pragma unroll
  for (int g = 0; g < 8; ++g) { mst[g] = -1e30f; lst[g] = 0.f; }

  const int nblk = (j0 + 16 + 31) >> 5;  // i-blocks of 32 keys; last col = j0+15
  for (int ib = 0; ib < nblk; ++ib) {
    const int i0 = ib << 5;
    const __bf16* kb0 = qk + ((size_t)bh * 2048 + i0) * 64;
    f32x8 s0 = f32x8{}, s1 = f32x8{};
    s0 = wmma_bf16(aq0, load_frag_b(kb0,            64, lane), s0);
    s0 = wmma_bf16(aq1, load_frag_b(kb0 + 32,       64, lane), s0);
    s1 = wmma_bf16(aq0, load_frag_b(kb0 + 16 * 64,      64, lane), s1);
    s1 = wmma_bf16(aq1, load_frag_b(kb0 + 16 * 64 + 32, 64, lane), s1);

#pragma unroll
    for (int g = 0; g < 8; ++g) {
      const int j = j0 + g + half * 8;              // this lane's row
      const bool k0ok = (i0 + nn) <= j;
      const bool k1ok = (i0 + 16 + nn) <= j;
      float v0 = k0ok ? s0[g] : -1e30f;
      float v1 = k1ok ? s1[g] : -1e30f;
      float mx = fmaxf(v0, v1);
#pragma unroll
      for (int o = 1; o < 16; o <<= 1) mx = fmaxf(mx, __shfl_xor(mx, o, 32));
      const float mnew = fmaxf(mst[g], mx);
      const float p0 = k0ok ? exp2f(s0[g] - mnew) : 0.f;
      const float p1 = k1ok ? exp2f(s1[g] - mnew) : 0.f;
      float rs = p0 + p1;
#pragma unroll
      for (int o = 1; o < 16; o <<= 1) rs += __shfl_xor(rs, o, 32);
      const float scale = exp2f(mst[g] - mnew);
      lst[g] = lst[g] * scale + rs;
      mst[g] = mnew;
#pragma unroll
      for (int t = 0; t < 4; ++t) accO[t][g] *= scale;
      sP[wave][g + half * 8][nn]      = (__bf16)p0;
      sP[wave][g + half * 8][16 + nn] = (__bf16)p1;
    }

    const bf16x16 ap = load_frag_a(&sP[wave][0][0], 32, lane);
#pragma unroll
    for (int t = 0; t < 4; ++t) {
      const __bf16* vb = vt + ((size_t)bh * 64 + t * 16) * 2048 + i0;
      accO[t] = wmma_bf16(ap, load_frag_b(vb, 2048, lane), accO[t]);
    }
  }

#pragma unroll
  for (int t = 0; t < 4; ++t)
#pragma unroll
    for (int g = 0; g < 8; ++g) {
      const int j = j0 + g + half * 8;
      const float y = accO[t][g] / lst[g];
      const int d = t * 16 + nn;
      ybuf[((size_t)(b * 2048 + j)) * 1024 + d * 16 + h] = (__bf16)y;
    }
}

// ---------------- launch ----------------
extern "C" void kernel_launch(void* const* d_in, const int* in_sizes, int n_in,
                              void* d_out, int out_size, void* d_ws, size_t ws_size,
                              hipStream_t stream) {
  const float* x      = (const float*)d_in[0];
  const float* w_attn = (const float*)d_in[1];
  const float* b_attn = (const float*)d_in[2];
  const float* w_proj = (const float*)d_in[3];
  const float* b_proj = (const float*)d_in[4];
  float* out = (float*)d_out;

  char* ws = (char*)d_ws;
  size_t off = 0;
  auto alloc = [&](size_t bytes) -> void* {
    void* p = ws + off;
    off += (bytes + 255) & ~(size_t)255;
    return p;
  };
  __bf16* xb  = (__bf16*)alloc((size_t)8192 * 1024 * 2);
  __bf16* wat = (__bf16*)alloc((size_t)3072 * 1024 * 2);
  __bf16* wpt = (__bf16*)alloc((size_t)1024 * 1024 * 2);
  __bf16* qk  = (__bf16*)alloc((size_t)64 * 2048 * 64 * 2);
  __bf16* kq  = (__bf16*)alloc((size_t)64 * 2048 * 64 * 2);
  __bf16* vt  = (__bf16*)alloc((size_t)64 * 2048 * 64 * 2);
  __bf16* yb  = (__bf16*)alloc((size_t)8192 * 1024 * 2);

  f32_to_bf16<<<(8388608 + 255) / 256, 256, 0, stream>>>(x, xb, 8388608);
  f32_to_bf16_t<<<(3145728 + 255) / 256, 256, 0, stream>>>(w_attn, wat, 1024, 3072);
  f32_to_bf16_t<<<(1048576 + 255) / 256, 256, 0, stream>>>(w_proj, wpt, 1024, 1024);

  // qkv GEMM: M=8192, N=3072, K=1024 -> scatter epilogue
  gemm_wmma<<<(8192 / 128) * (3072 / 128), 256, 0, stream>>>(
      xb, wat, b_attn, nullptr, qk, kq, vt, 8192, 3072, 1024, 1);

  // causal flash attention: 64 (b,h) x 16 j-tiles of 128
  flash_attn<<<64 * 16, 256, 0, stream>>>(qk, kq, vt, yb);

  // output projection: M=8192, N=1024, K=1024 -> fp32 out
  gemm_wmma<<<(8192 / 128) * (1024 / 128), 256, 0, stream>>>(
      yb, wpt, b_proj, out, nullptr, nullptr, nullptr, 8192, 1024, 1024, 0);
}
